// PhiLayer_22643067585104
// MI455X (gfx1250) — compile-verified, matrix operand order checked by reading, not run on hardware
//
#include <hip/hip_runtime.h>

// ---------------------------------------------------------------------------
// Phi-style transformer layer for MI455X (gfx1250, wave32, WMMA bf16 + TDM)
// B=2, S=2048, D=2560, H=32, DH=80, RD=32, NI=10240
// ---------------------------------------------------------------------------

#define B_   2
#define S_   2048
#define D_   2560
#define H_   32
#define DH_  80
#define NI_  10240
#define M_   (B_ * S_)      // 4096 rows
#define ND3_ (3 * D_)       // 7680

typedef unsigned int u32;
typedef __attribute__((ext_vector_type(16))) __bf16          v16bf;
typedef __attribute__((ext_vector_type(8)))  float           v8f;
typedef __attribute__((ext_vector_type(8)))  unsigned short  ushort8;
typedef __attribute__((ext_vector_type(4)))  unsigned int    u32x4;
typedef __attribute__((ext_vector_type(8)))  int             i32x8;
typedef __attribute__((ext_vector_type(4)))  int             i32x4;

union FragU { ushort8 h[2]; v16bf v; };

__device__ __forceinline__ unsigned short f2bf(float f) {
  unsigned int u = __float_as_uint(f);
  u += 0x7fffu + ((u >> 16) & 1u);            // round to nearest even
  return (unsigned short)(u >> 16);
}

// Load one 16-bit A/B WMMA fragment from LDS following the CDNA5 half-lane
// layout: lanes 0-15 hold K=[0..7]+[16..23], lanes 16-31 hold K=[8..15]+[24..31].
__device__ __forceinline__ v16bf ld_frag(const unsigned short* p, int kbase, int half) {
  FragU f;
  f.h[0] = *reinterpret_cast<const ushort8*>(p + kbase + half * 8);
  f.h[1] = *reinterpret_cast<const ushort8*>(p + kbase + 16 + half * 8);
  return f.v;
}

__device__ __forceinline__ float gelu_tanh(float x) {
  float x3 = x * x * x;
  return 0.5f * x * (1.0f + tanhf(0.7978845608f * (x + 0.044715f * x3)));
}

// ---------------------------------------------------------------------------
// Tensor Data Mover: 2-D tile DMA global -> LDS (bf16 elements).
// Builds the D# descriptor per CDNA5 ISA ch.8:
//   group0: count=1 | lds_addr | global_addr(57b) | type=2
//   group1: data_size=2B, optional LDS padding (16 DW interval, 4 DW pad
//           -> reproduces a 40-element padded row stride for a 32-elem row),
//           tensor_dim0/1, tile_dim0/1, tensor_dim0_stride
//   groups2/3: zero (2-D tensor)
// Issue from ONE wave only; completion via s_wait_tensorcnt.
// ---------------------------------------------------------------------------
__device__ __forceinline__ void tdm_load_2d_bf16(
    u32 lds_off, const void* gsrc,
    u32 tensor_d0, u32 tensor_d1,          // tensor extents (elements)
    u32 tile_d0, u32 tile_d1,              // tile extents (elements)
    unsigned long long stride0,            // dim0 stride (elements)
    bool pad_rows)                         // pad each 16 DW with 4 DW in LDS
{
  unsigned long long ga = (unsigned long long)(uintptr_t)gsrc;
  u32x4 g0;
  g0.x = 1u;                                   // count=1 (valid user descriptor)
  g0.y = lds_off;                              // lds_addr (bytes)
  g0.z = (u32)ga;                              // global_addr[31:0]
  g0.w = (u32)(ga >> 32) | (2u << 30);         // global_addr[56:32] | type=2

  u32 flags = (1u << 16);                      // data_size = 1 -> 2 bytes
  if (pad_rows) flags |= (1u << 20)            // pad_enable
               | (3u << 22)                    // pad_interval: 16 DWORDs
               | (3u << 25);                   // pad_amount:   4 DWORDs
  i32x8 g1;
  g1[0] = (int)flags;
  g1[1] = (int)((tensor_d0 & 0xffffu) << 16);                    // tensor_dim0[15:0]
  g1[2] = (int)((tensor_d0 >> 16) | ((tensor_d1 & 0xffffu) << 16));
  g1[3] = (int)((tensor_d1 >> 16) | (tile_d0 << 16));            // tile_dim0
  g1[4] = (int)(tile_d1 & 0xffffu);                              // tile_dim1 (tile_dim2=0)
  g1[5] = (int)(u32)(stride0 & 0xffffffffull);                   // dim0_stride[31:0]
  g1[6] = (int)(u32)(stride0 >> 32);                             // dim0_stride[47:32]
  g1[7] = 0;

  i32x4 z4 = {0, 0, 0, 0};
#if __clang_major__ >= 23
  i32x8 z8 = {0, 0, 0, 0, 0, 0, 0, 0};
  __builtin_amdgcn_tensor_load_to_lds(g0, g1, z4, z4, z8, 0);
#else
  __builtin_amdgcn_tensor_load_to_lds(g0, g1, z4, z4, 0);
#endif
}

// ---------------------------------------------------------------------------
// f32 -> bf16 cast (weights)
// ---------------------------------------------------------------------------
__global__ void k_cast_bf16(const float* __restrict__ src, unsigned short* __restrict__ dst, size_t n) {
  size_t i = (size_t)blockIdx.x * blockDim.x + threadIdx.x;
  size_t stride = (size_t)gridDim.x * blockDim.x;
  for (; i < n; i += stride) dst[i] = f2bf(src[i]);
}

// ---------------------------------------------------------------------------
// residual copy d_out = hidden
// ---------------------------------------------------------------------------
__global__ void k_copy_f32(const float* __restrict__ src, float* __restrict__ dst, size_t n) {
  size_t i = (size_t)blockIdx.x * blockDim.x + threadIdx.x;
  size_t stride = (size_t)gridDim.x * blockDim.x;
  for (; i < n; i += stride) dst[i] = src[i];
}

// ---------------------------------------------------------------------------
// LayerNorm (one 256-thread block per row) -> bf16 x
// ---------------------------------------------------------------------------
__global__ void __launch_bounds__(256) k_layernorm(
    const float* __restrict__ hs, const float* __restrict__ g, const float* __restrict__ b,
    unsigned short* __restrict__ xb) {
  const int row = blockIdx.x;
  const int t = threadIdx.x;
  const float* hr = hs + (size_t)row * D_;

  float s = 0.f, s2 = 0.f;
  for (int d = t; d < D_; d += 256) { float v = hr[d]; s += v; s2 += v * v; }
  #pragma unroll
  for (int m = 16; m >= 1; m >>= 1) { s += __shfl_xor(s, m, 32); s2 += __shfl_xor(s2, m, 32); }

  __shared__ float red0[8], red1[8], stats[2];
  const int wid = t >> 5, lid = t & 31;
  if (lid == 0) { red0[wid] = s; red1[wid] = s2; }
  __syncthreads();
  if (t < 8) {
    s = red0[t]; s2 = red1[t];
    #pragma unroll
    for (int m = 4; m >= 1; m >>= 1) { s += __shfl_xor(s, m, 8); s2 += __shfl_xor(s2, m, 8); }
    if (t == 0) {
      float mu = s / (float)D_;
      float var = s2 / (float)D_ - mu * mu;
      stats[0] = mu;
      stats[1] = rsqrtf(var + 1e-5f);
    }
  }
  __syncthreads();
  const float mu = stats[0], rstd = stats[1];
  for (int d = t; d < D_; d += 256)
    xb[(size_t)row * D_ + d] = f2bf((hr[d] - mu) * rstd * g[d] + b[d]);
}

// ---------------------------------------------------------------------------
// bf16 WMMA GEMM: C[M,N] = A[M,K] @ B[K,N] + bias, 128x128x32 tiles, 8 waves.
// A tile staged by the Tensor Data Mover (wave 0 issues, s_wait_tensorcnt);
// B tile staged transposed by all waves (WMMA B-fragment needs n-major LDS).
// MODE 0: store f32; MODE 1: gelu -> store bf16; MODE 2: add into f32.
// ---------------------------------------------------------------------------
#define BM 128
#define BN 128
#define BK 32
#define LDT 40   // padded LDS row stride (bf16) == 16 DW + 4 DW TDM padding

template <int MODE>
__global__ void __launch_bounds__(256) k_gemm_bf16(
    const unsigned short* __restrict__ A, const unsigned short* __restrict__ Bw,
    const float* __restrict__ bias, float* __restrict__ Cf, unsigned short* __restrict__ Cb,
    int M, int N, int K) {
  __shared__ unsigned short As[BM][LDT];   // row-major tile (TDM-filled, padded)
  __shared__ unsigned short Bs[BN][LDT];   // transposed tile (n-major)

  const int t = threadIdx.x;
  const int wave = t >> 5, lane = t & 31;
  const int wm = wave >> 2, wn = wave & 3;           // 2 x 4 wave grid
  const int half = lane >> 4, l16 = lane & 15;
  const int m0 = blockIdx.y * BM, n0 = blockIdx.x * BN;
  const u32 asOff = (u32)(uintptr_t)&As[0][0];       // LDS byte offset of As

  v8f acc[4][2] = {};

  for (int k0 = 0; k0 < K; k0 += BK) {
    __syncthreads();                       // prior compute done before overwrite

    // --- A tile via Tensor Data Mover: 128 x 32 bf16, padded rows in LDS ---
    if (wave == 0) {
      tdm_load_2d_bf16(asOff, A + (size_t)m0 * K + k0,
                       (u32)(K - k0), (u32)(M - m0),   // tensor extents from tile origin
                       BK, BM, (unsigned long long)K, true);
    }

    // --- B tile transposed: global [K,N] row-major -> LDS [n][k] ---
    #pragma unroll
    for (int i = 0; i < 2; ++i) {
      int c = t + i * 256;
      int kr = c >> 4, nc = (c & 15) * 8;
      const unsigned short* src = &Bw[(size_t)(k0 + kr) * N + n0 + nc];
      __builtin_prefetch(src + (size_t)BK * N, 0, 0);  // next K tile
      ushort8 vld = *reinterpret_cast<const ushort8*>(src);
      #pragma unroll
      for (int j = 0; j < 8; ++j) Bs[nc + j][kr] = vld[j];
    }

    if (wave == 0) __builtin_amdgcn_s_wait_tensorcnt(0);
    __syncthreads();

    v16bf af[4], bf[2];
    #pragma unroll
    for (int i = 0; i < 4; ++i) af[i] = ld_frag(&As[wm * 64 + i * 16 + l16][0], 0, half);
    #pragma unroll
    for (int j = 0; j < 2; ++j) bf[j] = ld_frag(&Bs[wn * 32 + j * 16 + l16][0], 0, half);

    #pragma unroll
    for (int i = 0; i < 4; ++i)
      #pragma unroll
      for (int j = 0; j < 2; ++j)
        acc[i][j] = __builtin_amdgcn_wmma_f32_16x16x32_bf16(
            false, af[i], false, bf[j], (short)0, acc[i][j], false, false);
  }

  // --- epilogue: C layout VGPR r -> row = r + 8*half, col = lane&15 ---
  #pragma unroll
  for (int i = 0; i < 4; ++i) {
    #pragma unroll
    for (int j = 0; j < 2; ++j) {
      const int col = n0 + wn * 32 + j * 16 + l16;
      const float bv = bias[col];
      #pragma unroll
      for (int r = 0; r < 8; ++r) {
        const int row = m0 + wm * 64 + i * 16 + r + 8 * half;
        const size_t idx = (size_t)row * N + col;
        float v = acc[i][j][r] + bv;
        if (MODE == 0)      Cf[idx] = v;
        else if (MODE == 1) Cb[idx] = f2bf(gelu_tanh(v));
        else                Cf[idx] += v;
      }
    }
  }
}

// ---------------------------------------------------------------------------
// Partial RoPE (NeoX, RD=32) on qkv f32; emit head-major bf16 Q/K/V [b,h,s,80]
// ---------------------------------------------------------------------------
__global__ void __launch_bounds__(256) k_rope(
    const float* __restrict__ qkv, unsigned short* __restrict__ Qd,
    unsigned short* __restrict__ Kd, unsigned short* __restrict__ Vd) {
  const int row = blockIdx.x;               // b*S + s
  const int s = row & (S_ - 1);
  const int b = row >> 11;
  const float* base = qkv + (size_t)row * ND3_;
  for (int n = threadIdx.x; n < D_; n += 256) {
    const int h = n / DH_, dh = n % DH_;
    const size_t dst = ((size_t)(b * H_ + h) * S_ + s) * DH_ + dh;
    float qv = base[n], kv = base[D_ + n], vv = base[2 * D_ + n];
    if (dh < 32) {
      const int i = dh & 15;
      const float inv = __powf(10000.f, -(float)i / 16.f);
      float sn, cs;
      __sincosf((float)s * inv, &sn, &cs);
      if (dh < 16) {
        float q2 = base[n + 16], k2 = base[D_ + n + 16];
        qv = qv * cs - q2 * sn;
        kv = kv * cs - k2 * sn;
      } else {
        float q1 = base[n - 16], k1 = base[D_ + n - 16];
        qv = qv * cs + q1 * sn;
        kv = kv * cs + k1 * sn;
      }
    }
    Qd[dst] = f2bf(qv); Kd[dst] = f2bf(kv); Vd[dst] = f2bf(vv);
  }
}

// ---------------------------------------------------------------------------
// Flash attention: 64-q tile / (b,h); 4 waves (16 q rows each); 32-key steps.
// Q@K^T and P@V via WMMA bf16 (dh padded 80->96). Online softmax in f32.
// ---------------------------------------------------------------------------
__global__ void __launch_bounds__(128) k_attention(
    const unsigned short* __restrict__ Q, const unsigned short* __restrict__ Kt,
    const unsigned short* __restrict__ Vt, unsigned short* __restrict__ O) {
  __shared__ unsigned short Qs[64][96];
  __shared__ unsigned short Ks[32][96];
  __shared__ unsigned short Vs[96][32];
  __shared__ unsigned short Ps[4][16][32];

  const int qt = blockIdx.x;                  // q tile of 64
  const int bh = blockIdx.y;                  // b*H + h
  const int b = bh >> 5, h = bh & 31;
  const int t = threadIdx.x, wave = t >> 5, lane = t & 31;
  const int half = lane >> 4, l16 = lane & 15;
  const float scale = 0.11180339887f;         // 80^-0.5
  const size_t headOff = (size_t)bh * S_ * DH_;

  for (int i = t; i < 64 * 96; i += 128) {
    int r = i / 96, c = i % 96;
    Qs[r][c] = (c < DH_) ? Q[headOff + (size_t)(qt * 64 + r) * DH_ + c] : (unsigned short)0;
  }
  __syncthreads();

  v16bf qf[3];
  #pragma unroll
  for (int ks = 0; ks < 3; ++ks) qf[ks] = ld_frag(&Qs[wave * 16 + l16][0], ks * 32, half);

  float mrow[8], lrow[8];
  v8f acc[5] = {};
  #pragma unroll
  for (int r = 0; r < 8; ++r) { mrow[r] = -3.0e38f; lrow[r] = 0.f; }

  const int nkt = (qt + 1) * 2;               // causal: keys <= qTile end
  for (int kt = 0; kt < nkt; ++kt) {
    __syncthreads();                          // prior-iteration K/V reads done
    for (int i = t; i < 32 * 96; i += 128) {
      int r = i / 96, c = i % 96;
      Ks[r][c] = (c < DH_) ? Kt[headOff + (size_t)(kt * 32 + r) * DH_ + c] : (unsigned short)0;
    }
    for (int i = t; i < 96 * 32; i += 128) {
      int dh = i / 32, kk = i % 32;
      Vs[dh][kk] = (dh < DH_) ? Vt[headOff + (size_t)(kt * 32 + kk) * DH_ + dh] : (unsigned short)0;
    }
    __syncthreads();

    // S = Q @ K^T over padded dh (3 x K=32 steps), two 16-key fragments
    v8f sc[2] = {};
    #pragma unroll
    for (int nf = 0; nf < 2; ++nf)
      #pragma unroll
      for (int ks = 0; ks < 3; ++ks) {
        v16bf kf = ld_frag(&Ks[nf * 16 + l16][0], ks * 32, half);
        sc[nf] = __builtin_amdgcn_wmma_f32_16x16x32_bf16(
            false, qf[ks], false, kf, (short)0, sc[nf], false, false);
      }

    // causal mask + scale + row max
    float rmax[8];
    #pragma unroll
    for (int r = 0; r < 8; ++r) rmax[r] = -3.0e38f;
    #pragma unroll
    for (int nf = 0; nf < 2; ++nf)
      #pragma unroll
      for (int r = 0; r < 8; ++r) {
        const int qrow = qt * 64 + wave * 16 + r + 8 * half;
        const int key = kt * 32 + nf * 16 + l16;
        float v = (key <= qrow) ? sc[nf][r] * scale : -3.0e38f;
        sc[nf][r] = v;
        rmax[r] = fmaxf(rmax[r], v);
      }
    #pragma unroll
    for (int m = 1; m <= 8; m <<= 1)
      #pragma unroll
      for (int r = 0; r < 8; ++r) rmax[r] = fmaxf(rmax[r], __shfl_xor(rmax[r], m, 32));

    // online softmax update
    float corr[8], rsum[8];
    #pragma unroll
    for (int r = 0; r < 8; ++r) {
      float mn = fmaxf(mrow[r], rmax[r]);
      corr[r] = __expf(mrow[r] - mn);
      mrow[r] = mn;
      rsum[r] = 0.f;
    }
    #pragma unroll
    for (int nf = 0; nf < 2; ++nf)
      #pragma unroll
      for (int r = 0; r < 8; ++r) {
        float p = __expf(sc[nf][r] - mrow[r]);
        sc[nf][r] = p;
        rsum[r] += p;
      }
    #pragma unroll
    for (int m = 1; m <= 8; m <<= 1)
      #pragma unroll
      for (int r = 0; r < 8; ++r) rsum[r] += __shfl_xor(rsum[r], m, 32);
    #pragma unroll
    for (int r = 0; r < 8; ++r) lrow[r] = lrow[r] * corr[r] + rsum[r];
    #pragma unroll
    for (int g = 0; g < 5; ++g)
      #pragma unroll
      for (int r = 0; r < 8; ++r) acc[g][r] *= corr[r];

    // stage P (C layout) -> per-wave LDS -> reload as A-fragment
    #pragma unroll
    for (int nf = 0; nf < 2; ++nf)
      #pragma unroll
      for (int r = 0; r < 8; ++r)
        Ps[wave][r + 8 * half][nf * 16 + l16] = f2bf(sc[nf][r]);
    __asm__ volatile("s_wait_dscnt 0" ::: "memory");

    v16bf pf = ld_frag(&Ps[wave][l16][0], 0, half);
    #pragma unroll
    for (int g = 0; g < 5; ++g) {
      v16bf vf = ld_frag(&Vs[g * 16 + l16][0], 0, half);
      acc[g] = __builtin_amdgcn_wmma_f32_16x16x32_bf16(
          false, pf, false, vf, (short)0, acc[g], false, false);
    }
  }

  // epilogue: O[b*S + q][h*80 + dh] bf16
  #pragma unroll
  for (int g = 0; g < 5; ++g)
    #pragma unroll
    for (int r = 0; r < 8; ++r) {
      const int qrow = qt * 64 + wave * 16 + r + 8 * half;
      const int dh = g * 16 + l16;
      O[((size_t)(b * S_) + qrow) * D_ + h * DH_ + dh] = f2bf(acc[g][r] / lrow[r]);
    }
}

// ---------------------------------------------------------------------------
// launcher
// ---------------------------------------------------------------------------
extern "C" void kernel_launch(void* const* d_in, const int* in_sizes, int n_in,
                              void* d_out, int out_size, void* d_ws, size_t ws_size,
                              hipStream_t stream) {
  (void)in_sizes; (void)n_in; (void)out_size; (void)ws_size;
  const float* hidden = (const float*)d_in[1];
  const float* ln_g   = (const float*)d_in[2];
  const float* ln_b   = (const float*)d_in[3];
  const float* wqkv   = (const float*)d_in[4];
  const float* bqkv   = (const float*)d_in[5];
  const float* wout   = (const float*)d_in[6];
  const float* bout   = (const float*)d_in[7];
  const float* wfc1   = (const float*)d_in[8];
  const float* bfc1   = (const float*)d_in[9];
  const float* wfc2   = (const float*)d_in[10];
  const float* bfc2   = (const float*)d_in[11];
  float* out = (float*)d_out;

  // workspace layout (all sizes 256B-aligned)
  char* p = (char*)d_ws;
  unsigned short* xb    = (unsigned short*)p; p += (size_t)M_ * D_ * 2;       // 20 MB
  unsigned short* wqkvb = (unsigned short*)p; p += (size_t)D_ * ND3_ * 2;     // 39 MB
  unsigned short* woutb = (unsigned short*)p; p += (size_t)D_ * D_ * 2;       // 13 MB
  unsigned short* wfc1b = (unsigned short*)p; p += (size_t)D_ * NI_ * 2;      // 52 MB
  unsigned short* wfc2b = (unsigned short*)p; p += (size_t)NI_ * D_ * 2;      // 52 MB
  float*          qkvf  = (float*)p;          p += (size_t)M_ * ND3_ * 4;     // 126 MB (reused as h1)
  unsigned short* qb    = (unsigned short*)p; p += (size_t)M_ * D_ * 2;
  unsigned short* kb    = (unsigned short*)p; p += (size_t)M_ * D_ * 2;
  unsigned short* vb    = (unsigned short*)p; p += (size_t)M_ * D_ * 2;
  unsigned short* attnb = (unsigned short*)p; p += (size_t)M_ * D_ * 2;
  unsigned short* h1b   = (unsigned short*)qkvf;   // alias: qkv f32 dead after RoPE

  // 1) weights -> bf16
  k_cast_bf16<<<4096, 256, 0, stream>>>(wqkv, wqkvb, (size_t)D_ * ND3_);
  k_cast_bf16<<<4096, 256, 0, stream>>>(wout, woutb, (size_t)D_ * D_);
  k_cast_bf16<<<4096, 256, 0, stream>>>(wfc1, wfc1b, (size_t)D_ * NI_);
  k_cast_bf16<<<4096, 256, 0, stream>>>(wfc2, wfc2b, (size_t)NI_ * D_);

  // 2) LayerNorm -> x (bf16)
  k_layernorm<<<M_, 256, 0, stream>>>(hidden, ln_g, ln_b, xb);

  // 3) qkv = x @ wqkv + bqkv   (f32 out)
  k_gemm_bf16<0><<<dim3(ND3_ / BN, M_ / BM), 256, 0, stream>>>(
      xb, wqkvb, bqkv, qkvf, nullptr, M_, ND3_, D_);

  // 4) RoPE -> head-major bf16 q/k/v
  k_rope<<<M_, 256, 0, stream>>>(qkvf, qb, kb, vb);

  // 5) flash attention -> attn (bf16, [b*S+s][D])
  k_attention<<<dim3(S_ / 64, B_ * H_), 128, 0, stream>>>(qb, kb, vb, attnb);

  // 6) out = residual
  k_copy_f32<<<4096, 256, 0, stream>>>(hidden, out, (size_t)M_ * D_);

  // 7) out += attn @ wout + bout
  k_gemm_bf16<2><<<dim3(D_ / BN, M_ / BM), 256, 0, stream>>>(
      attnb, woutb, bout, out, nullptr, M_, D_, D_);

  // 8) h1 = gelu(x @ wfc1 + bfc1)   (bf16 out, reuses qkv buffer)
  k_gemm_bf16<1><<<dim3(NI_ / BN, M_ / BM), 256, 0, stream>>>(
      xb, wfc1b, bfc1, nullptr, h1b, M_, NI_, D_);

  // 9) out += h1 @ wfc2 + bfc2
  k_gemm_bf16<2><<<dim3(D_ / BN, M_ / BM), 256, 0, stream>>>(
      h1b, wfc2b, bfc2, out, nullptr, M_, D_, NI_);
}
